// DKBATNet_22883585753703
// MI455X (gfx1250) — matrix operand order repaired and checked by reading, not compiled, not run on hardware
//
#include <hip/hip_runtime.h>
#include <hip/hip_bf16.h>

// ---------------- problem constants (from reference) ----------------
#define N_NODES 50000
#define N_EDGES 400000
#define N_REL   500
#define XS      100      // X_SIZE
#define GS      100      // G_SIZE
#define NEG     0.01f    // leaky slope
#define EPSN    1e-12f

// K paddings (multiples of 32 for bf16 WMMA K-steps)
#define KPAD1 320        // 2*100 + 100 -> 320
#define KPAD2 384        // 2*128 + 100 -> 384

// edge groups (64 edges each) processed per block between weight stagings
#define EDGE_ITERS 5     // 400000 = 1250 blocks * 5 * 64

typedef __bf16 bf16;
typedef bf16  v8bf  __attribute__((ext_vector_type(8)));
typedef bf16  v16bf __attribute__((ext_vector_type(16)));
typedef float v8f   __attribute__((ext_vector_type(8)));

__device__ __forceinline__ float leaky(float x) { return x > 0.f ? x : NEG * x; }

// Async global->LDS bulk copy path (CDNA5 GLOBAL_LOAD_ASYNC_TO_LDS_B128,
// tracked by ASYNCcnt). Guarded: falls back to plain vector copies if the
// builtin is not declared by this toolchain. Probe round 2 showed the builtin
// exists and takes typed int4 pointers (global src, LDS dst).
#if defined(__HIP_DEVICE_COMPILE__) && __has_builtin(__builtin_amdgcn_global_load_async_to_lds_b128)
#define ASYNC_LDS_COPY 1
typedef int int4v __attribute__((vector_size(16)));
typedef __attribute__((address_space(1))) int4v* gint4p;
typedef __attribute__((address_space(3))) int4v* lint4p;
#endif

// ---------------------------------------------------------------
// Prep: build fused bf16 weight matrices B[n][k] = W^T with the
// "out" direction's src/dst column blocks swapped, plus bf16 g.
// B1: [256][KPAD1], B2: [256][KPAD2]; rows 0-127 = W_in, 128-255 = W_out(swapped)
// ---------------------------------------------------------------
__global__ void prep_kernel(const float* __restrict__ W1i, const float* __restrict__ W1o,
                            const float* __restrict__ W2i, const float* __restrict__ W2o,
                            const float* __restrict__ g,
                            bf16* __restrict__ B1, bf16* __restrict__ B2,
                            bf16* __restrict__ gbf) {
    int stride = gridDim.x * blockDim.x;
    int tid = blockIdx.x * blockDim.x + threadIdx.x;
    for (int i = tid; i < 256 * KPAD1; i += stride) {
        int n = i / KPAD1, k = i % KPAD1;
        float v = 0.f;
        if (k < 300) {
            if (n < 128) v = W1i[n * 300 + k];
            else {
                int ks = (k < 100) ? k + 100 : ((k < 200) ? k - 100 : k);
                v = W1o[(n - 128) * 300 + ks];
            }
        }
        B1[i] = (bf16)v;
    }
    for (int i = tid; i < 256 * KPAD2; i += stride) {
        int n = i / KPAD2, k = i % KPAD2;
        float v = 0.f;
        if (k < 356) {
            if (n < 128) v = W2i[n * 356 + k];
            else {
                int ks = (k < 128) ? k + 128 : ((k < 256) ? k - 128 : k);
                v = W2o[(n - 128) * 356 + ks];
            }
        }
        B2[i] = (bf16)v;
    }
    for (int i = tid; i < N_REL * GS; i += stride) gbf[i] = (bf16)g[i];
}

// ---------------------------------------------------------------
// x row-normalize -> bf16 features; entity linear ent = xn @ We^T + be
// one wave per node (8 nodes / 256-thread block)
// ---------------------------------------------------------------
__global__ void norm_entity_kernel(const float* __restrict__ x,
                                   const float* __restrict__ We, const float* __restrict__ be,
                                   bf16* __restrict__ xnbf, float* __restrict__ ent) {
    __shared__ float xs[8][XS];
    int wave = threadIdx.x >> 5, lane = threadIdx.x & 31;
    int node = blockIdx.x * 8 + wave;
    bool act = node < N_NODES;
    float ss = 0.f;
    if (act) {
        for (int k = lane; k < XS; k += 32) {
            float v = x[(size_t)node * XS + k];
            xs[wave][k] = v;
            ss += v * v;
        }
    }
#pragma unroll
    for (int m = 1; m < 32; m <<= 1) ss += __shfl_xor(ss, m, 32);
    float inv = 1.f / fmaxf(sqrtf(ss), EPSN);
    __syncthreads();
    if (act) {
        for (int k = lane; k < XS; k += 32)
            xnbf[(size_t)node * XS + k] = (bf16)(xs[wave][k] * inv);
        for (int o = lane; o < 64; o += 32) {
            float s = 0.f;
            for (int k = 0; k < XS; k++) s += We[o * XS + k] * xs[wave][k];
            ent[(size_t)node * 64 + o] = s * inv + be[o];
        }
    }
}

// ---------------------------------------------------------------
// Edge attention kernel (both passes).
//   Block: 256 threads = 8 waves; stages fused weights into LDS ONCE,
//   then processes EDGE_ITERS groups of 64 edges against them.
//   Wave w: edge-tile (w&3), direction (w>>2): 0=in (cols 0..127), 1=out (128..255)
//   C[16 edges x 128 cols] per wave via v_wmma_f32_16x16x32_bf16, K-loop over Kpad.
//   pass 0: logits a -> e=exp(leaky(a)) -> atomicAdd softmax denominators
//   pass 1: alpha = e/denom; scatter-add 0.5*alpha*c into node accumulator
// ---------------------------------------------------------------
__global__ __launch_bounds__(256)
void edge_attn_kernel(const bf16* __restrict__ nodefeat, int F,
                      const bf16* __restrict__ gbf,
                      const int* __restrict__ eidx, const int* __restrict__ etype,
                      const bf16* __restrict__ Bw, int Kpad,
                      const float* __restrict__ att_in, const float* __restrict__ att_out,
                      const float* __restrict__ b_in, const float* __restrict__ b_out,
                      float* __restrict__ denom, float* __restrict__ hacc,
                      int pass) {
    extern __shared__ char smem[];
    bf16* Bsh   = (bf16*)smem;                                 // [256][Kpad]
    bf16* feats = (bf16*)(smem + (size_t)256 * Kpad * 2);      // [64][Kpad]
    int*  rsh   = (int*)(smem + (size_t)256 * Kpad * 2 + (size_t)64 * Kpad * 2);
    int*  csh   = rsh + 64;

    int tid = threadIdx.x;

    // ---- stage fused weights into LDS once per block ----
    {
        const uint4* Bg = (const uint4*)Bw;
        uint4* Bs = (uint4*)Bsh;
        int nB = (256 * Kpad * 2) >> 4;
#ifdef ASYNC_LDS_COPY
        for (int i = tid; i < nB; i += 256) {
            __builtin_amdgcn_global_load_async_to_lds_b128(
                (gint4p)(Bg + i), (lint4p)(Bs + i), 0, 0);
        }
#if __has_builtin(__builtin_amdgcn_s_wait_asynccnt)
        __builtin_amdgcn_s_wait_asynccnt(0);
#else
        asm volatile("s_wait_asynccnt 0" ::: "memory");
#endif
#else
        for (int i = tid; i < nB; i += 256) Bs[i] = Bg[i];
#endif
    }

    int wave = tid >> 5, lane = tid & 31;
    int tile = wave & 3, dir = wave >> 2;
    int l16 = lane & 15, half = lane >> 4;

    const float* attv = dir ? att_out : att_in;   // flat [2*64] == column index
    const float* bv   = dir ? b_out   : b_in;
    float attr[8], biasr[8];
#pragma unroll
    for (int nt = 0; nt < 8; nt++) {
        attr[nt]  = attv[nt * 16 + l16];
        biasr[nt] = bv[nt * 16 + l16];
    }

    for (int it = 0; it < EDGE_ITERS; ++it) {
        int eb = (blockIdx.x * EDGE_ITERS + it) * 64;

        // ---- gather this group's edge indices + features ----
        if (tid < 64) {
            rsh[tid] = eidx[eb + tid];
            csh[tid] = eidx[N_EDGES + eb + tid];
        }
        {
            int e = tid >> 2, j = tid & 3;
            int src = eidx[eb + e];
            int dst = eidx[N_EDGES + eb + e];
            int et  = etype[eb + e];
            unsigned* frow = (unsigned*)(feats + (size_t)e * Kpad);
            const unsigned* srow = (const unsigned*)(nodefeat + (size_t)src * F);
            const unsigned* drow = (const unsigned*)(nodefeat + (size_t)dst * F);
            const unsigned* grow = (const unsigned*)(gbf + (size_t)et * GS);
            int hF = F >> 1, hG = GS >> 1;
            int nu = Kpad >> 1;
            for (int u = j; u < nu; u += 4) {
                unsigned v = 0u;
                if      (u < hF)          v = srow[u];
                else if (u < 2 * hF)      v = drow[u - hF];
                else if (u < 2 * hF + hG) v = grow[u - 2 * hF];
                frow[u] = v;
            }
        }
        __syncthreads();   // feats (and, first time, Bsh) visible to all waves

        // ---- WMMA: C[16 x 128] per wave ----
        v8f acc[8];
#pragma unroll
        for (int nt = 0; nt < 8; nt++) acc[nt] = (v8f){0.f,0.f,0.f,0.f,0.f,0.f,0.f,0.f};

        // A frag: lane = row M, K halves at +8*half / +16+8*half (ISA 16-bit A layout)
        // B frag: lane = col N, K halves at +8*half / +16+8*half (same order as A)
        const bf16* arow = feats + (size_t)(tile * 16 + l16) * Kpad + 8 * half;
        const bf16* brow = Bsh + (size_t)(dir * 128 + l16) * Kpad + 8 * half;

        for (int k0 = 0; k0 < Kpad; k0 += 32) {
            v8bf a0 = *(const v8bf*)(arow + k0);
            v8bf a1 = *(const v8bf*)(arow + k0 + 16);
            v16bf af = __builtin_shufflevector(a0, a1, 0,1,2,3,4,5,6,7,8,9,10,11,12,13,14,15);
#pragma unroll
            for (int nt = 0; nt < 8; nt++) {
                const bf16* bb = brow + (size_t)nt * 16 * Kpad + k0;
                v8bf b0 = *(const v8bf*)(bb);
                v8bf b1 = *(const v8bf*)(bb + 16);
                v16bf bm = __builtin_shufflevector(b0, b1, 0,1,2,3,4,5,6,7,8,9,10,11,12,13,14,15);
                acc[nt] = __builtin_amdgcn_wmma_f32_16x16x32_bf16(
                    false, af, false, bm, (short)0, acc[nt], false, false);
            }
        }
        // + bias (part of c)
#pragma unroll
        for (int nt = 0; nt < 8; nt++)
#pragma unroll
            for (int r = 0; r < 8; r++) acc[nt][r] += biasr[nt];

        // attention logits: per edge-row, per head; cross-lane reduce in 16-lane halves
        float ev[2][8];
#pragma unroll
        for (int h = 0; h < 2; h++)
#pragma unroll
            for (int r = 0; r < 8; r++) {
                float t = 0.f;
#pragma unroll
                for (int q = 0; q < 4; q++) t += attr[h * 4 + q] * acc[h * 4 + q][r];
#pragma unroll
                for (int m = 1; m < 16; m <<= 1) t += __shfl_xor(t, m, 32);
                ev[h][r] = expf(leaky(t));
            }

        int rowbase = tile * 16 + 8 * half;   // edge row within group for vgpr index r
        if (pass == 0) {
            if (l16 == 0) {
#pragma unroll
                for (int r = 0; r < 8; r++) {
                    int er = rowbase + r;
                    int seg = dir ? csh[er] : rsh[er];   // in: segment by row; out: by col
                    atomicAdd(&denom[(size_t)dir * (2 * N_NODES) + (size_t)seg * 2 + 0], ev[0][r]);
                    atomicAdd(&denom[(size_t)dir * (2 * N_NODES) + (size_t)seg * 2 + 1], ev[1][r]);
                }
            }
        } else {
            float al[2][8];
#pragma unroll
            for (int r = 0; r < 8; r++) {
                int er = rowbase + r;
                int seg = dir ? csh[er] : rsh[er];
                al[0][r] = ev[0][r] / denom[(size_t)dir * (2 * N_NODES) + (size_t)seg * 2 + 0];
                al[1][r] = ev[1][r] / denom[(size_t)dir * (2 * N_NODES) + (size_t)seg * 2 + 1];
            }
#pragma unroll
            for (int r = 0; r < 8; r++) {
                int er = rowbase + r;
                int tgt = dir ? rsh[er] : csh[er];       // in: scatter to col; out: to row
                float* base = hacc + (size_t)tgt * 128 + l16;
#pragma unroll
                for (int nt = 0; nt < 8; nt++) {
                    float v = 0.5f * al[nt >> 2][r] * acc[nt][r];   // ALPHA = 0.5 both dirs
                    atomicAdd(base + nt * 16, v);
                }
            }
        }
        __syncthreads();   // protect feats/rsh/csh before next group's gather
    }
}

// ---------------------------------------------------------------
// Node finish: leaky + per-head l2norm.
//  mode 0: write bf16 features for layer 2
//  mode 1: + entity broadcast, final l2norm over 128 -> d_out
// one wave per node.
// ---------------------------------------------------------------
__global__ void node_finish_kernel(const float* __restrict__ hacc,
                                   const float* __restrict__ ent,
                                   bf16* __restrict__ h1bf,
                                   float* __restrict__ outp,
                                   int mode) {
    int wave = threadIdx.x >> 5, lane = threadIdx.x & 31;
    int node = blockIdx.x * 8 + wave;
    if (node >= N_NODES) return;
    float v[4];
#pragma unroll
    for (int i = 0; i < 4; i++) {
        int c = lane + 32 * i;
        v[i] = leaky(hacc[(size_t)node * 128 + c]);
    }
    float s0 = v[0] * v[0] + v[1] * v[1];     // head 0: c < 64
    float s1 = v[2] * v[2] + v[3] * v[3];     // head 1: c >= 64
#pragma unroll
    for (int m = 1; m < 32; m <<= 1) { s0 += __shfl_xor(s0, m, 32); s1 += __shfl_xor(s1, m, 32); }
    float i0 = 1.f / fmaxf(sqrtf(s0), EPSN);
    float i1 = 1.f / fmaxf(sqrtf(s1), EPSN);
    v[0] *= i0; v[1] *= i0; v[2] *= i1; v[3] *= i1;
    if (mode == 0) {
#pragma unroll
        for (int i = 0; i < 4; i++)
            h1bf[(size_t)node * 128 + lane + 32 * i] = (bf16)v[i];
    } else {
        float w[4], ss = 0.f;
#pragma unroll
        for (int i = 0; i < 4; i++) {
            int c = lane + 32 * i;
            w[i] = v[i] + ent[(size_t)node * 64 + (c & 63)];
            ss += w[i] * w[i];
        }
#pragma unroll
        for (int m = 1; m < 32; m <<= 1) ss += __shfl_xor(ss, m, 32);
        float inv = 1.f / fmaxf(sqrtf(ss), EPSN);
#pragma unroll
        for (int i = 0; i < 4; i++)
            outp[(size_t)node * 128 + lane + 32 * i] = w[i] * inv;
    }
}

// ---------------------------------------------------------------
// g' = g @ Wr^T + br  (tiny: 500 x 128 x 100)
// ---------------------------------------------------------------
__global__ void gprime_kernel(const float* __restrict__ g, const float* __restrict__ Wr,
                              const float* __restrict__ br, float* __restrict__ out2) {
    int idx = blockIdx.x * blockDim.x + threadIdx.x;
    if (idx >= N_REL * 128) return;
    int r = idx >> 7, o = idx & 127;
    float s = br[o];
    for (int k = 0; k < GS; k++) s += Wr[o * GS + k] * g[(size_t)r * GS + k];
    out2[idx] = s;
}

// ---------------------------------------------------------------
extern "C" void kernel_launch(void* const* d_in, const int* in_sizes, int n_in,
                              void* d_out, int out_size, void* d_ws, size_t ws_size,
                              hipStream_t stream) {
    (void)in_sizes; (void)n_in; (void)out_size; (void)ws_size;
    const float* x    = (const float*)d_in[0];
    const float* g    = (const float*)d_in[1];
    const int*   eidx = (const int*)d_in[2];
    const int*   etyp = (const int*)d_in[3];
    const float* W1i  = (const float*)d_in[4];
    const float* b1i  = (const float*)d_in[5];
    const float* a1i  = (const float*)d_in[6];
    const float* W1o  = (const float*)d_in[7];
    const float* b1o  = (const float*)d_in[8];
    const float* a1o  = (const float*)d_in[9];
    const float* W2i  = (const float*)d_in[10];
    const float* b2i  = (const float*)d_in[11];
    const float* a2i  = (const float*)d_in[12];
    const float* W2o  = (const float*)d_in[13];
    const float* b2o  = (const float*)d_in[14];
    const float* a2o  = (const float*)d_in[15];
    const float* We   = (const float*)d_in[16];
    const float* be   = (const float*)d_in[17];
    const float* Wr   = (const float*)d_in[18];
    const float* br   = (const float*)d_in[19];
    float* out = (float*)d_out;

    // workspace carve (~63 MB total)
    char* w = (char*)d_ws;
    auto carve = [&](size_t bytes) -> char* {
        char* p = w; w += (bytes + 255) & ~(size_t)255; return p;
    };
    bf16*  xnbf = (bf16*)carve((size_t)N_NODES * XS * 2);
    bf16*  h1bf = (bf16*)carve((size_t)N_NODES * 128 * 2);
    bf16*  gbf  = (bf16*)carve((size_t)N_REL * GS * 2);
    bf16*  B1   = (bf16*)carve((size_t)256 * KPAD1 * 2);
    bf16*  B2   = (bf16*)carve((size_t)256 * KPAD2 * 2);
    float* ent  = (float*)carve((size_t)N_NODES * 64 * 4);
    float* den  = (float*)carve((size_t)2 * N_NODES * 2 * 4);
    float* hacc = (float*)carve((size_t)N_NODES * 128 * 4);

    int eblocks = N_EDGES / (64 * EDGE_ITERS);                // 1250
    int nblocks = (N_NODES + 7) / 8;                          // 6250
    size_t sh1 = (size_t)(256 + 64) * KPAD1 * 2 + 512;        // ~205 KB (< 320 KB WGP LDS)
    size_t sh2 = (size_t)(256 + 64) * KPAD2 * 2 + 512;        // ~246 KB

    (void)hipMemsetAsync(den, 0, (size_t)2 * N_NODES * 2 * 4, stream);
    (void)hipMemsetAsync(hacc, 0, (size_t)N_NODES * 128 * 4, stream);

    prep_kernel<<<512, 256, 0, stream>>>(W1i, W1o, W2i, W2o, g, B1, B2, gbf);
    norm_entity_kernel<<<nblocks, 256, 0, stream>>>(x, We, be, xnbf, ent);

    // layer 1
    edge_attn_kernel<<<eblocks, 256, sh1, stream>>>(xnbf, XS, gbf, eidx, etyp, B1, KPAD1,
                                                    a1i, a1o, b1i, b1o, den, hacc, 0);
    edge_attn_kernel<<<eblocks, 256, sh1, stream>>>(xnbf, XS, gbf, eidx, etyp, B1, KPAD1,
                                                    a1i, a1o, b1i, b1o, den, hacc, 1);
    node_finish_kernel<<<nblocks, 256, 0, stream>>>(hacc, ent, h1bf, out, 0);

    // layer 2
    (void)hipMemsetAsync(den, 0, (size_t)2 * N_NODES * 2 * 4, stream);
    (void)hipMemsetAsync(hacc, 0, (size_t)N_NODES * 128 * 4, stream);
    edge_attn_kernel<<<eblocks, 256, sh2, stream>>>(h1bf, 128, gbf, eidx, etyp, B2, KPAD2,
                                                    a2i, a2o, b2i, b2o, den, hacc, 0);
    edge_attn_kernel<<<eblocks, 256, sh2, stream>>>(h1bf, 128, gbf, eidx, etyp, B2, KPAD2,
                                                    a2i, a2o, b2i, b2o, den, hacc, 1);
    node_finish_kernel<<<nblocks, 256, 0, stream>>>(hacc, ent, h1bf, out, 1);

    gprime_kernel<<<(N_REL * 128 + 255) / 256, 256, 0, stream>>>(g, Wr, br,
                                                                 out + (size_t)N_NODES * 128);
}
